// GraphTransformer_16295105921340
// MI455X (gfx1250) — matrix operand order repaired and checked by reading, not compiled, not run on hardware
//
#include <hip/hip_runtime.h>
#include <cmath>

// ---------------------------------------------------------------------------
// GraphTransformer (2x TransformerConv + BN/ReLU + log_softmax) for MI455X.
//   * Layer-1 QKVS projections: f16 WMMA (v_wmma_f32_16x16x32_f16). One block
//     per 16-row x tile; A-fragments built once and reused across the 4 weight
//     matrices (16 WMMAs/wave). Weight tiles (128x128 f16) are DMA'd into a
//     double-buffered LDS region by the Tensor Data Mover (tensor_load_to_lds,
//     s_wait_tensorcnt) while the previous matrix is being computed.
//   * Edge segment-softmax: ordered-int atomic max + global_atomic_add_f32.
//   * BN stats: one block per channel, LDS tree reduction.
//   * Layer-2 (128->2): VALU GEMV with all 4 weight mats in LDS.
// ---------------------------------------------------------------------------

#define NN   50000
#define EE   800000
#define INC  128
#define HIDC 32
#define H1C  4
#define HCC  128
#define OUTC 2
#define EPSV 1e-5f

typedef __attribute__((ext_vector_type(16))) _Float16 v16h;
typedef __attribute__((ext_vector_type(8)))  _Float16 v8h;
typedef __attribute__((ext_vector_type(8)))  float    v8f;
typedef __attribute__((ext_vector_type(4)))  unsigned int v4u;
typedef __attribute__((ext_vector_type(8)))  unsigned int v8u;

__device__ __forceinline__ void atomicMaxF(float* addr, float val) {
    // Ordered-int trick (init = -inf): positives compare as signed ints,
    // negatives reverse-order as unsigned.
    if (val >= 0.0f) atomicMax((int*)addr, __float_as_int(val));
    else             atomicMin((unsigned int*)addr, __float_as_uint(val));
}

// ---- TDM: DMA a rows x cols tile of f16 (row stride = cols) into LDS -------
// D# layout per CDNA5 ISA (2D tensor: groups 0 and 1 only).
__device__ __forceinline__ void tdm_load_f16_tile(const _Float16* gptr,
                                                  unsigned lds_byte_addr,
                                                  unsigned rows, unsigned cols) {
    const unsigned long long ga = (unsigned long long)(uintptr_t)gptr;
    v4u g0;
    g0[0] = 1u;                                         // count=1, user descriptor
    g0[1] = lds_byte_addr;                              // [63:32]  lds_addr
    g0[2] = (unsigned)(ga & 0xffffffffu);               // [95:64]  global_addr lo
    g0[3] = (unsigned)((ga >> 32) & 0x01ffffffu)        // [120:96] global_addr hi
          | (2u << 30);                                 // [127:126] type = 2
    v8u g1;
    g1[0] = 1u << 16;                                   // data_size=1 (2 bytes)
    g1[1] = (cols & 0xffffu) << 16;                     // tensor_dim0[15:0]
    g1[2] = ((cols >> 16) & 0xffffu)                    // tensor_dim0[31:16]
          | ((rows & 0xffffu) << 16);                   // tensor_dim1[15:0]
    g1[3] = (cols & 0xffffu) << 16;                     // tile_dim0
    g1[4] = rows & 0xffffu;                             // tile_dim1 (tile_dim2=0)
    g1[5] = cols;                                       // tensor_dim0_stride lo32
    g1[6] = 0u;                                         // stride hi / dim1_stride lo
    g1[7] = 0u;                                         // dim1_stride hi
    asm volatile("tensor_load_to_lds %0, %1" :: "s"(g0), "s"(g1) : "memory");
}

// ---- weight pre-transpose + f32->f16 : wt[mat][col(HC)][k(IN)] --------------
__global__ __launch_bounds__(256)
void gt_w1_transpose(const float* __restrict__ Wq, const float* __restrict__ Wk,
                     const float* __restrict__ Wv, const float* __restrict__ Ws,
                     _Float16* __restrict__ wt) {
    int idx = blockIdx.x * 256 + threadIdx.x;
    if (idx >= 4 * HCC * INC) return;
    int mat = idx / (HCC * INC);
    int rem = idx - mat * (HCC * INC);
    int col = rem / INC;
    int k   = rem - col * INC;
    const float* W = (mat == 0) ? Wq : (mat == 1) ? Wk : (mat == 2) ? Wv : Ws;
    wt[idx] = (_Float16)W[(size_t)k * HCC + col];
}

// ---- layer-1 QKVS projection: WMMA + TDM double-buffered weights ------------
// grid = N/16 blocks, 256 threads (8 waves). Wave w owns output cols [16w,16w+16).
__global__ __launch_bounds__(256)
void gt_qkvs_wmma(const float* __restrict__ x, const _Float16* __restrict__ wt,
                  const float* __restrict__ bq, const float* __restrict__ bk,
                  const float* __restrict__ bv, const float* __restrict__ bs,
                  float* __restrict__ oq, float* __restrict__ ok,
                  float* __restrict__ ov, float* __restrict__ os) {
    __shared__ _Float16 xs[16 * INC];            // 4 KB x tile (f16)
    __shared__ _Float16 wb[2][HCC * INC];        // 2 x 32 KB weight tiles (TDM dest)
    const int tid  = threadIdx.x;
    const int lane = tid & 31;
    const int wave = tid >> 5;
    const int row0 = blockIdx.x << 4;

    // Kick the TDM for matrix 0 into buffer 0 (wave 0 owns TENSORcnt).
    if (wave == 0)
        tdm_load_f16_tile(wt, (unsigned)(uintptr_t)&wb[0][0], HCC, INC);

    // Cooperative f32 -> f16 staging of the 16 x 128 x tile.
    for (int t = tid; t < 16 * INC; t += 256)
        xs[t] = (_Float16)x[(size_t)(row0 + (t >> 7)) * INC + (t & 127)];
    __syncthreads();

    const int n    = lane & 15;      // A row / output col within tile
    const int lh   = lane >> 4;      // lane half
    const int col0 = wave << 4;

    // A-fragments are shared by all 4 weight matrices: build once.
    // A k order used here (contiguous, not the ISA table order): a[0..15] = K = kk+16*lh+0..15 (matches B).
    v16h afrag[4];
    #pragma unroll
    for (int t = 0; t < 4; ++t) {
        const int kk = t << 5;
        const v8h lo = *reinterpret_cast<const v8h*>(&xs[n * INC + kk + (lh << 4)]);
        const v8h hi = *reinterpret_cast<const v8h*>(&xs[n * INC + kk + 8 + (lh << 4)]);
        v16h a;
        #pragma unroll
        for (int i = 0; i < 8; ++i) { a[i] = lo[i]; a[8 + i] = hi[i]; }
        afrag[t] = a;
    }

    #pragma unroll
    for (int mat = 0; mat < 4; ++mat) {
        if (wave == 0) __builtin_amdgcn_s_wait_tensorcnt(0);  // wb[mat&1] landed
        __syncthreads();                                      // publish to all waves
        if (wave == 0 && mat < 3)                             // prefetch next matrix
            tdm_load_f16_tile(wt + (size_t)(mat + 1) * (HCC * INC),
                              (unsigned)(uintptr_t)&wb[(mat + 1) & 1][0], HCC, INC);

        const _Float16* wl = wb[mat & 1];
        v8f acc = {};
        #pragma unroll
        for (int t = 0; t < 4; ++t) {
            // B k order used here (contiguous, same as A): b[0..15] = K = kk + 16*lh + 0..15,
            // contiguous in the [col][k] staged tile -> one 32B LDS read.
            const v16h b = *reinterpret_cast<const v16h*>(
                &wl[(col0 + n) * INC + (t << 5) + (lh << 4)]);
            acc = __builtin_amdgcn_wmma_f32_16x16x32_f16(false, afrag[t], false, b,
                                                         (short)0, acc, false, false);
        }

        const float* bias = (mat == 0) ? bq : (mat == 1) ? bk : (mat == 2) ? bv : bs;
        float*       out  = (mat == 0) ? oq : (mat == 1) ? ok : (mat == 2) ? ov : os;
        #pragma unroll
        for (int r = 0; r < 8; ++r) {
            const int m = r + (lh << 3);
            out[(size_t)(row0 + m) * HCC + col0 + n] = acc[r] + bias[col0 + n];
        }
        __syncthreads();   // all waves done with wb[mat&1] before TDM reuses it
    }
}

// ---- segment-softmax scratch init ------------------------------------------
__global__ __launch_bounds__(256)
void gt_init_seg(float* __restrict__ mx, float* __restrict__ den, int count) {
    int i = blockIdx.x * 256 + threadIdx.x;
    if (i < count) { mx[i] = -__builtin_inff(); den[i] = 0.0f; }
}

// ---- layer-1 edge logits + segment max (one thread per edge*head) ----------
__global__ __launch_bounds__(256)
void gt_edge_logits1(const int* __restrict__ ei, const float* __restrict__ q,
                     const float* __restrict__ k, float* __restrict__ alpha,
                     float* __restrict__ mx) {
    int idx = blockIdx.x * 256 + threadIdx.x;
    if (idx >= EE * H1C) return;
    const int e = idx >> 2, h = idx & 3;
    const int src = ei[e], dst = ei[EE + e];
    const float4* qp = (const float4*)(q + (size_t)dst * HCC + h * HIDC);
    const float4* kp = (const float4*)(k + (size_t)src * HCC + h * HIDC);
    float s = 0.0f;
    #pragma unroll
    for (int i = 0; i < 8; ++i) {
        float4 a = qp[i], b = kp[i];
        s += a.x * b.x + a.y * b.y + a.z * b.z + a.w * b.w;
    }
    s *= 0.17677669529663687f;                      // 1/sqrt(32)
    alpha[idx] = s;
    atomicMaxF(&mx[(size_t)dst * H1C + h], s);
}

// ---- layer-1 exp + segment sum ---------------------------------------------
__global__ __launch_bounds__(256)
void gt_edge_expsum1(const int* __restrict__ ei, const float* __restrict__ mx,
                     float* __restrict__ alpha, float* __restrict__ den) {
    int idx = blockIdx.x * 256 + threadIdx.x;
    if (idx >= EE * H1C) return;
    const int e = idx >> 2, h = idx & 3;
    const int dst = ei[EE + e];
    const float w = expf(alpha[idx] - mx[(size_t)dst * H1C + h]);
    alpha[idx] = w;
    atomicAdd(&den[(size_t)dst * H1C + h], w);
}

// ---- layer-1 weighted scatter (one thread per edge*channel) ----------------
__global__ __launch_bounds__(256)
void gt_edge_scatter1(const int* __restrict__ ei, const float* __restrict__ alpha,
                      const float* __restrict__ den, const float* __restrict__ v,
                      float* __restrict__ out) {
    int idx = blockIdx.x * 256 + threadIdx.x;
    if (idx >= EE * HCC) return;
    const int e = idx >> 7, c = idx & 127, h = c >> 5;
    const int src = ei[e], dst = ei[EE + e];
    const float w = alpha[e * H1C + h] / den[(size_t)dst * H1C + h];
    atomicAdd(&out[(size_t)dst * HCC + c], w * v[(size_t)src * HCC + c]);
}

// ---- BatchNorm statistics (one block per channel) --------------------------
__global__ __launch_bounds__(256)
void gt_bn_stats(const float* __restrict__ h, float* __restrict__ mu,
                 float* __restrict__ rstd) {
    __shared__ float s1[256], s2[256];
    const int c = blockIdx.x, tid = threadIdx.x;
    float a = 0.0f, b = 0.0f;
    for (int r = tid; r < NN; r += 256) {
        const float v = h[(size_t)r * HCC + c];
        a += v; b += v * v;
    }
    s1[tid] = a; s2[tid] = b; __syncthreads();
    for (int s = 128; s > 0; s >>= 1) {
        if (tid < s) { s1[tid] += s1[tid + s]; s2[tid] += s2[tid + s]; }
        __syncthreads();
    }
    if (tid == 0) {
        const float m   = s1[0] / (float)NN;
        const float var = s2[0] / (float)NN - m * m;
        mu[c] = m; rstd[c] = rsqrtf(var + EPSV);
    }
}

// ---- BN apply + ReLU --------------------------------------------------------
__global__ __launch_bounds__(256)
void gt_bn_apply(const float* __restrict__ x, const float* __restrict__ mu,
                 const float* __restrict__ rstd, const float* __restrict__ g,
                 const float* __restrict__ b, float* __restrict__ out) {
    int idx = blockIdx.x * 256 + threadIdx.x;
    if (idx >= NN * HCC) return;
    const int c = idx & 127;
    const float v = g[c] * (x[idx] - mu[c]) * rstd[c] + b[c];
    out[idx] = v > 0.0f ? v : 0.0f;
}

// ---- layer-2 projections (128 -> 2, x4 mats; weights in LDS) ---------------
__global__ __launch_bounds__(256)
void gt_l2_linear(const float* __restrict__ h,
                  const float* __restrict__ W2q, const float* __restrict__ W2k,
                  const float* __restrict__ W2v, const float* __restrict__ W2s,
                  const float* __restrict__ b2q, const float* __restrict__ b2k,
                  const float* __restrict__ b2v, const float* __restrict__ b2s,
                  float* __restrict__ q2, float* __restrict__ k2,
                  float* __restrict__ v2, float* __restrict__ o2) {
    __shared__ float sw[4][HCC * OUTC];
    for (int t = threadIdx.x; t < HCC * OUTC; t += 256) {
        sw[0][t] = W2q[t]; sw[1][t] = W2k[t]; sw[2][t] = W2v[t]; sw[3][t] = W2s[t];
    }
    __syncthreads();
    const int n = blockIdx.x * 256 + threadIdx.x;
    if (n >= NN) return;
    float acc[8] = { b2q[0], b2q[1], b2k[0], b2k[1],
                     b2v[0], b2v[1], b2s[0], b2s[1] };
    const float* hr = h + (size_t)n * HCC;
    #pragma unroll 4
    for (int c = 0; c < HCC; ++c) {
        const float hv = hr[c];
        acc[0] += hv * sw[0][c * 2];  acc[1] += hv * sw[0][c * 2 + 1];
        acc[2] += hv * sw[1][c * 2];  acc[3] += hv * sw[1][c * 2 + 1];
        acc[4] += hv * sw[2][c * 2];  acc[5] += hv * sw[2][c * 2 + 1];
        acc[6] += hv * sw[3][c * 2];  acc[7] += hv * sw[3][c * 2 + 1];
    }
    q2[n * 2] = acc[0]; q2[n * 2 + 1] = acc[1];
    k2[n * 2] = acc[2]; k2[n * 2 + 1] = acc[3];
    v2[n * 2] = acc[4]; v2[n * 2 + 1] = acc[5];
    o2[n * 2] = acc[6]; o2[n * 2 + 1] = acc[7];   // skip term initializes out2
}

// ---- layer-2 edge logits + segment max -------------------------------------
__global__ __launch_bounds__(256)
void gt_edge_logits2(const int* __restrict__ ei, const float* __restrict__ q2,
                     const float* __restrict__ k2, float* __restrict__ alpha,
                     float* __restrict__ mx) {
    int e = blockIdx.x * 256 + threadIdx.x;
    if (e >= EE) return;
    const int src = ei[e], dst = ei[EE + e];
    const float s = (q2[dst * 2] * k2[src * 2] + q2[dst * 2 + 1] * k2[src * 2 + 1])
                    * 0.7071067811865476f;          // 1/sqrt(2)
    alpha[e] = s;
    atomicMaxF(&mx[dst], s);
}

__global__ __launch_bounds__(256)
void gt_edge_expsum2(const int* __restrict__ ei, const float* __restrict__ mx,
                     float* __restrict__ alpha, float* __restrict__ den) {
    int e = blockIdx.x * 256 + threadIdx.x;
    if (e >= EE) return;
    const int dst = ei[EE + e];
    const float w = expf(alpha[e] - mx[dst]);
    alpha[e] = w;
    atomicAdd(&den[dst], w);
}

__global__ __launch_bounds__(256)
void gt_edge_scatter2(const int* __restrict__ ei, const float* __restrict__ alpha,
                      const float* __restrict__ den, const float* __restrict__ v2,
                      float* __restrict__ out) {
    int idx = blockIdx.x * 256 + threadIdx.x;
    if (idx >= EE * OUTC) return;
    const int e = idx >> 1, c = idx & 1;
    const int src = ei[e], dst = ei[EE + e];
    const float w = alpha[e] / den[dst];
    atomicAdd(&out[dst * 2 + c], w * v2[src * 2 + c]);
}

// ---- final log_softmax over 2 classes --------------------------------------
__global__ __launch_bounds__(256)
void gt_logsoftmax(const float* __restrict__ h2, float* __restrict__ out) {
    int n = blockIdx.x * 256 + threadIdx.x;
    if (n >= NN) return;
    const float a = h2[n * 2], b = h2[n * 2 + 1];
    const float m = fmaxf(a, b);
    const float lse = m + logf(expf(a - m) + expf(b - m));
    out[n * 2] = a - lse;
    out[n * 2 + 1] = b - lse;
}

// ---------------------------------------------------------------------------
extern "C" void kernel_launch(void* const* d_in, const int* in_sizes, int n_in,
                              void* d_out, int out_size, void* d_ws, size_t ws_size,
                              hipStream_t stream) {
    (void)in_sizes; (void)n_in; (void)out_size; (void)ws_size;
    const float* x   = (const float*)d_in[0];
    const int*   ei  = (const int*)d_in[1];
    const float* W1q = (const float*)d_in[2];  const float* b1q = (const float*)d_in[3];
    const float* W1k = (const float*)d_in[4];  const float* b1k = (const float*)d_in[5];
    const float* W1v = (const float*)d_in[6];  const float* b1v = (const float*)d_in[7];
    const float* W1s = (const float*)d_in[8];  const float* b1s = (const float*)d_in[9];
    const float* bng = (const float*)d_in[10]; const float* bnb = (const float*)d_in[11];
    const float* W2q = (const float*)d_in[12]; const float* b2q = (const float*)d_in[13];
    const float* W2k = (const float*)d_in[14]; const float* b2k = (const float*)d_in[15];
    const float* W2v = (const float*)d_in[16]; const float* b2v = (const float*)d_in[17];
    const float* W2s = (const float*)d_in[18]; const float* b2s = (const float*)d_in[19];
    float* out = (float*)d_out;

    // workspace carve (256B aligned bump allocator)
    char* p = (char*)d_ws;
    auto alloc = [&](size_t bytes) -> char* {
        char* r = p; p += (bytes + 255) & ~(size_t)255; return r;
    };
    _Float16* wt   = (_Float16*)alloc(4 * HCC * INC * sizeof(_Float16));
    float* q1      = (float*)alloc((size_t)NN * HCC * sizeof(float));
    float* k1      = (float*)alloc((size_t)NN * HCC * sizeof(float));
    float* v1      = (float*)alloc((size_t)NN * HCC * sizeof(float));
    float* out1    = (float*)alloc((size_t)NN * HCC * sizeof(float)); // s + attn
    float* alpha1  = (float*)alloc((size_t)EE * H1C * sizeof(float));
    float* m1      = (float*)alloc((size_t)NN * H1C * sizeof(float));
    float* den1    = (float*)alloc((size_t)NN * H1C * sizeof(float));
    float* mu      = (float*)alloc(HCC * sizeof(float));
    float* rstd    = (float*)alloc(HCC * sizeof(float));
    float* hbuf    = (float*)alloc((size_t)NN * HCC * sizeof(float));
    float* q2      = (float*)alloc((size_t)NN * OUTC * sizeof(float));
    float* k2      = (float*)alloc((size_t)NN * OUTC * sizeof(float));
    float* v2      = (float*)alloc((size_t)NN * OUTC * sizeof(float));
    float* out2    = (float*)alloc((size_t)NN * OUTC * sizeof(float));
    float* alpha2  = (float*)alloc((size_t)EE * sizeof(float));
    float* m2      = (float*)alloc((size_t)NN * sizeof(float));
    float* den2    = (float*)alloc((size_t)NN * sizeof(float));

    // --- layer 1 ---
    gt_w1_transpose<<<(4 * HCC * INC + 255) / 256, 256, 0, stream>>>(
        W1q, W1k, W1v, W1s, wt);
    gt_qkvs_wmma<<<NN / 16, 256, 0, stream>>>(
        x, wt, b1q, b1k, b1v, b1s, q1, k1, v1, out1);
    gt_init_seg<<<(NN * H1C + 255) / 256, 256, 0, stream>>>(m1, den1, NN * H1C);
    gt_edge_logits1<<<(EE * H1C + 255) / 256, 256, 0, stream>>>(ei, q1, k1, alpha1, m1);
    gt_edge_expsum1<<<(EE * H1C + 255) / 256, 256, 0, stream>>>(ei, m1, alpha1, den1);
    gt_edge_scatter1<<<(EE * HCC + 255) / 256, 256, 0, stream>>>(
        ei, alpha1, den1, v1, out1);

    // --- BN + ReLU ---
    gt_bn_stats<<<HCC, 256, 0, stream>>>(out1, mu, rstd);
    gt_bn_apply<<<(NN * HCC + 255) / 256, 256, 0, stream>>>(
        out1, mu, rstd, bng, bnb, hbuf);

    // --- layer 2 ---
    gt_l2_linear<<<(NN + 255) / 256, 256, 0, stream>>>(
        hbuf, W2q, W2k, W2v, W2s, b2q, b2k, b2v, b2s, q2, k2, v2, out2);
    gt_init_seg<<<(NN + 255) / 256, 256, 0, stream>>>(m2, den2, NN);
    gt_edge_logits2<<<(EE + 255) / 256, 256, 0, stream>>>(ei, q2, k2, alpha2, m2);
    gt_edge_expsum2<<<(EE + 255) / 256, 256, 0, stream>>>(ei, m2, alpha2, den2);
    gt_edge_scatter2<<<(EE * OUTC + 255) / 256, 256, 0, stream>>>(
        ei, alpha2, den2, v2, out2);

    gt_logsoftmax<<<(NN + 255) / 256, 256, 0, stream>>>(out2, out);
}
